// different_Self_Attention_27290222199364
// MI455X (gfx1250) — matrix operand order repaired and checked by reading, not compiled, not run on hardware
//
#include <hip/hip_runtime.h>
#include <hip/hip_bf16.h>

typedef __attribute__((ext_vector_type(16))) __bf16 bf16x16;
typedef __attribute__((ext_vector_type(8)))  float  f32x8;
typedef unsigned short u16;
typedef unsigned int   u32;

union FragB16 {
    bf16x16 v;
    uint4   u[2];
};

__constant__ float c_pw[41] = {
    0.1f,0.1f,0.1f,0.1f,0.1f,0.1f,0.1f,0.1f,0.1f,0.1f,0.1f,0.1f,0.1f,0.1f,
    0.63f,0.63f,0.76f,0.81f,0.82f,0.96f,0.1f,0.96f,0.82f,0.81f,0.76f,0.63f,0.63f,
    0.1f,0.1f,0.1f,0.1f,0.1f,0.1f,0.1f,0.1f,0.1f,0.1f,0.1f,0.1f,0.1f,0.1f
};

__device__ __forceinline__ u16 f2bf(float f) {
    u32 u = __float_as_uint(f);
    u32 r = u + 0x7FFFu + ((u >> 16) & 1u);   // round-to-nearest-even
    return (u16)(r >> 16);
}

// packed f32x2 -> bf16x2: native v_cvt_pk_bf16_f32 on gfx1250 (1 VALU op)
__device__ __forceinline__ u32 pack_bf16(float a, float b) {
#if defined(__gfx1250__)
    u32 r;
    asm("v_cvt_pk_bf16_f32 %0, %1, %2" : "=v"(r) : "v"(a), "v"(b));
    return r;
#else
    return (u32)f2bf(a) | ((u32)f2bf(b) << 16);
#endif
}

// single f32 -> bf16 via the packed converter (low half)
__device__ __forceinline__ u16 cvt_bf16(float a) {
#if defined(__gfx1250__)
    return (u16)pack_bf16(a, 0.f);
#else
    return f2bf(a);
#endif
}

// Async global->LDS copy of 16 bytes per lane (CDNA5, ASYNCcnt-tracked).
// GV mode: 64-bit address in VGPR pair, LDS byte address in a VGPR.
__device__ __forceinline__ void async_copy_b128(u32 lds_addr, const void* gaddr) {
#if defined(__gfx1250__)
    asm volatile("global_load_async_to_lds_b128 %0, %1, off"
                 :: "v"(lds_addr), "v"(gaddr) : "memory");
#endif
}

__device__ __forceinline__ void wait_asynccnt0() {
#if defined(__gfx1250__)
    asm volatile("s_wait_asynccnt 0x0" ::: "memory");
#endif
}

// ---------------------------------------------------------------------------
// Kernel 0: concat + convert weights (fp32 -> bf16) and biases (fp32 concat).
// Wcat is 3072x1024 bf16 (6 MB) -> resident in 192MB L2 for the GEMM.
// ---------------------------------------------------------------------------
__global__ __launch_bounds__(256) void convert_weights(
    const float* __restrict__ wq, const float* __restrict__ wk, const float* __restrict__ wv,
    const float* __restrict__ bq, const float* __restrict__ bk, const float* __restrict__ bv,
    u16* __restrict__ wcat, float* __restrict__ bcat)
{
    int idx = blockIdx.x * blockDim.x + threadIdx.x;
    int e0  = idx * 4;
    if (e0 < 3 * 1024 * 1024) {
        int w   = e0 >> 20;                 // which weight matrix
        int rem = e0 & 0xFFFFF;             // element within it
        const float* src = (w == 0) ? wq : (w == 1) ? wk : wv;
        float4 f = *(const float4*)(src + rem);
        *(uint2*)(wcat + e0) = make_uint2(pack_bf16(f.x, f.y), pack_bf16(f.z, f.w));
    }
    if (idx < 768) {                        // 768*4 = 3072 bias elements
        int b0 = idx * 4;
        for (int j = 0; j < 4; ++j) {
            int i  = b0 + j;
            int w2 = i >> 10;
            const float* bs = (w2 == 0) ? bq : (w2 == 1) ? bk : bv;
            bcat[i] = bs[i & 1023];
        }
    }
}

// ---------------------------------------------------------------------------
// Kernel 1: fused QKV projection GEMM.
//   M = 20992 tokens, N = 3072 (Q|K|V), K = 1024.
//   128x128 block tile, K-step 32, double-buffered LDS.
//   B tile: async global->LDS copy (ASYNCcnt). A tile: fp32->bf16 in VGPRs.
//   v_wmma_f32_16x16x32_bf16, f32 accum. Epilogue: pw[s]*(acc+bias) -> bf16.
// ---------------------------------------------------------------------------
#define BM 128
#define BN 128
#define BK 32
#define NKB (1024 / BK)

__global__ __launch_bounds__(256) void gemm_qkv(
    const float* __restrict__ X, const u16* __restrict__ Wc,
    const float* __restrict__ bcat, u16* __restrict__ qkv)
{
    __shared__ __align__(16) u16 lA[2][BM * BK];   // 2 x 8 KB
    __shared__ __align__(16) u16 lB[2][BN * BK];   // 2 x 8 KB

    const int t     = threadIdx.x;
    const int lane  = t & 31;
    const int wid   = t >> 5;
    const int waveM = wid & 3;                  // 4 waves over M (32 rows each)
    const int waveN = wid >> 2;                 // 2 waves over N (64 cols each)
    const int m0    = blockIdx.x * BM;
    const int n0    = blockIdx.y * BN;

    const int l15    = lane & 15;
    const int half8  = (lane >= 16) ? 8 : 0;
    const int half16 = (lane >= 16) ? 16 : 0;

    const int lrow = t >> 1;                    // 128 rows, 2 threads/row
    const int lseg = t & 1;                     // 16-element half of BK

    f32x8 acc[2][4];
    const f32x8 zero = {0.f,0.f,0.f,0.f,0.f,0.f,0.f,0.f};
    for (int i = 0; i < 2; ++i)
        for (int j = 0; j < 4; ++j) acc[i][j] = zero;

    // ---- staging helpers ----
    auto stageB = [&](int kb, int buf) {        // async bf16 copy, 32B/thread
        const u16* bsrc = Wc + (size_t)(n0 + lrow) * 1024 + kb * BK + lseg * 16;
        u32 ldsa = (u32)(uintptr_t)&lB[buf][lrow * BK + lseg * 16];
        async_copy_b128(ldsa,      bsrc);
        async_copy_b128(ldsa + 16, bsrc + 8);
    };
    auto stageA = [&](int kb, int buf) {        // fp32 load -> bf16 -> LDS
        const float* asrc = X + (size_t)(m0 + lrow) * 1024 + kb * BK + lseg * 16;
        float4 f0 = ((const float4*)asrc)[0];
        float4 f1 = ((const float4*)asrc)[1];
        float4 f2 = ((const float4*)asrc)[2];
        float4 f3 = ((const float4*)asrc)[3];
        uint4 pa0, pa1;
        pa0.x = pack_bf16(f0.x, f0.y);  pa0.y = pack_bf16(f0.z, f0.w);
        pa0.z = pack_bf16(f1.x, f1.y);  pa0.w = pack_bf16(f1.z, f1.w);
        pa1.x = pack_bf16(f2.x, f2.y);  pa1.y = pack_bf16(f2.z, f2.w);
        pa1.z = pack_bf16(f3.x, f3.y);  pa1.w = pack_bf16(f3.z, f3.w);
        uint4* adst = (uint4*)&lA[buf][lrow * BK + lseg * 16];
        adst[0] = pa0; adst[1] = pa1;
    };

    // ---- prologue: fill buffer 0 ----
    stageB(0, 0);
    stageA(0, 0);
    wait_asynccnt0();
    __syncthreads();

    for (int kb = 0; kb < NKB; ++kb) {
        const int cur = kb & 1;
        const int nxt = cur ^ 1;

        // kick off next tile's async B copy as early as possible
        if (kb + 1 < NKB) stageB(kb + 1, nxt);

        // ---- fragments from current buffer (ISA VGPR layouts) ----
        FragB16 afr[2];
        #pragma unroll
        for (int mt = 0; mt < 2; ++mt) {
            int row = waveM * 32 + mt * 16 + l15;
            afr[mt].u[0] = *(const uint4*)&lA[cur][row * BK + half8];      // K {0..7}/{8..15}
            afr[mt].u[1] = *(const uint4*)&lA[cur][row * BK + 16 + half8]; // K {16..23}/{24..31}
        }
        FragB16 bfr[4];
        #pragma unroll
        for (int nt = 0; nt < 4; ++nt) {
            int col = waveN * 64 + nt * 16 + l15;
            bfr[nt].u[0] = *(const uint4*)&lB[cur][col * BK + half8];      // K {0..7}/{8..15}
            bfr[nt].u[1] = *(const uint4*)&lB[cur][col * BK + 16 + half8];
        }

        // overlap A-convert of next tile with this tile's WMMAs
        if (kb + 1 < NKB) stageA(kb + 1, nxt);

        #pragma unroll
        for (int mt = 0; mt < 2; ++mt)
            #pragma unroll
            for (int nt = 0; nt < 4; ++nt)
                acc[mt][nt] = __builtin_amdgcn_wmma_f32_16x16x32_bf16(
                    false, afr[mt].v, false, bfr[nt].v,
                    (short)0, acc[mt][nt], false, false);

        if (kb + 1 < NKB) wait_asynccnt0();
        __syncthreads();
    }

    // ---- epilogue: pw[s] * (acc + bias) -> bf16 QKV ----
    #pragma unroll
    for (int mt = 0; mt < 2; ++mt)
        #pragma unroll
        for (int nt = 0; nt < 4; ++nt)
            #pragma unroll
            for (int v = 0; v < 8; ++v) {
                int r = m0 + waveM * 32 + mt * 16 + half8 + v;
                int c = n0 + waveN * 64 + nt * 16 + l15;
                float val = c_pw[r % 41] * (acc[mt][nt][v] + bcat[c]);
                qkv[(size_t)r * 3072 + c] = cvt_bf16(val);
            }
}

// ---------------------------------------------------------------------------
// Kernel 2: attention, one wave per (batch, head). 8192 blocks of 32.
//   S = (Q K^T)/8 via WMMA (48x48 padded, K=64), masked softmax in fp32,
//   O = P V via WMMA (keys padded to 64 with zero probs), fp32 store.
// ---------------------------------------------------------------------------
__global__ __launch_bounds__(32) void attention(
    const u16* __restrict__ qkv, float* __restrict__ out)
{
    __shared__ __align__(16) u16   sQ [48 * 64];
    __shared__ __align__(16) u16   sK [48 * 64];
    __shared__ __align__(16) u16   sVt[64 * 64];   // [dim][key], keys >=41 zero
    __shared__ __align__(16) u16   sP [48 * 64];   // probs, key cols >=41 zero
    __shared__ __align__(16) float sS [48 * 48];

    const int bh = blockIdx.x;
    const int b  = bh >> 4;
    const int h  = bh & 15;
    const int t  = threadIdx.x;

    const int l15    = t & 15;
    const int half8  = (t >= 16) ? 8 : 0;
    const int half16 = (t >= 16) ? 16 : 0;
    const uint4 zz = make_uint4(0, 0, 0, 0);

    // ---- load Q,K rows and V transposed into LDS (pad to 48/64) ----
    for (int s = t; s < 64; s += 32) {
        if (s < 48) {
            uint4* dq = (uint4*)&sQ[s * 64];
            uint4* dk = (uint4*)&sK[s * 64];
            if (s < 41) {
                const size_t tok = (size_t)(b * 41 + s) * 3072 + h * 64;
                const uint4* q = (const uint4*)(qkv + tok);
                const uint4* k = (const uint4*)(qkv + tok + 1024);
                #pragma unroll
                for (int i = 0; i < 8; ++i) { dq[i] = q[i]; dk[i] = k[i]; }
            } else {
                #pragma unroll
                for (int i = 0; i < 8; ++i) { dq[i] = zz; dk[i] = zz; }
            }
        }
        // sVt column s (key s)
        if (s < 41) {
            const size_t tok = (size_t)(b * 41 + s) * 3072 + h * 64 + 2048;
            union { uint4 u4[8]; u16 us[64]; } tmp;
            const uint4* v = (const uint4*)(qkv + tok);
            #pragma unroll
            for (int i = 0; i < 8; ++i) tmp.u4[i] = v[i];
            #pragma unroll
            for (int d = 0; d < 64; ++d) sVt[d * 64 + s] = tmp.us[d];
        } else {
            #pragma unroll
            for (int d = 0; d < 64; ++d) sVt[d * 64 + s] = 0;
        }
    }
    __syncthreads();

    // ---- S = Q K^T * 1/sqrt(64); mask key cols >= 41 ----
    const f32x8 zero = {0.f,0.f,0.f,0.f,0.f,0.f,0.f,0.f};
    for (int mt = 0; mt < 3; ++mt)
        for (int nt = 0; nt < 3; ++nt) {
            f32x8 acc = zero;
            #pragma unroll
            for (int ks = 0; ks < 2; ++ks) {
                const int k0 = ks * 32;
                FragB16 a, bb;
                int row = mt * 16 + l15;
                a.u[0]  = *(const uint4*)&sQ[row * 64 + k0 + half8];
                a.u[1]  = *(const uint4*)&sQ[row * 64 + k0 + 16 + half8];
                int col = nt * 16 + l15;
                bb.u[0] = *(const uint4*)&sK[col * 64 + k0 + half8];
                bb.u[1] = *(const uint4*)&sK[col * 64 + k0 + 16 + half8];
                acc = __builtin_amdgcn_wmma_f32_16x16x32_bf16(
                    false, a.v, false, bb.v, (short)0, acc, false, false);
            }
            #pragma unroll
            for (int v = 0; v < 8; ++v) {
                int r = mt * 16 + half8 + v;
                int c = nt * 16 + l15;
                sS[r * 48 + c] = (c < 41) ? acc[v] * 0.125f : -1e30f;
            }
        }
    __syncthreads();

    // ---- row softmax (fp32), repack probs as bf16, pad key cols to 64 ----
    for (int r = t; r < 48; r += 32) {
        float mx = -1e30f;
        for (int c = 0; c < 41; ++c) mx = fmaxf(mx, sS[r * 48 + c]);
        float sum = 0.f;
        for (int c = 0; c < 41; ++c) sum += __expf(sS[r * 48 + c] - mx);
        float inv = 1.0f / sum;
        for (int c = 0; c < 64; ++c)
            sP[r * 64 + c] = (c < 41) ? cvt_bf16(__expf(sS[r * 48 + c] - mx) * inv)
                                      : (u16)0;
    }
    __syncthreads();

    // ---- O = P V ----
    for (int mt = 0; mt < 3; ++mt)
        for (int nt = 0; nt < 4; ++nt) {
            f32x8 acc = zero;
            #pragma unroll
            for (int ks = 0; ks < 2; ++ks) {
                const int k0 = ks * 32;
                FragB16 a, bb;
                int row = mt * 16 + l15;
                a.u[0]  = *(const uint4*)&sP[row * 64 + k0 + half8];
                a.u[1]  = *(const uint4*)&sP[row * 64 + k0 + 16 + half8];
                int col = nt * 16 + l15;            // head-dim column
                bb.u[0] = *(const uint4*)&sVt[col * 64 + k0 + half8];
                bb.u[1] = *(const uint4*)&sVt[col * 64 + k0 + 16 + half8];
                acc = __builtin_amdgcn_wmma_f32_16x16x32_bf16(
                    false, a.v, false, bb.v, (short)0, acc, false, false);
            }
            #pragma unroll
            for (int v = 0; v < 8; ++v) {
                int r = mt * 16 + half8 + v;
                int c = nt * 16 + l15;
                if (r < 41)
                    out[(size_t)(b * 41 + r) * 1024 + h * 64 + c] = acc[v];
            }
        }
}

// ---------------------------------------------------------------------------
extern "C" void kernel_launch(void* const* d_in, const int* in_sizes, int n_in,
                              void* d_out, int out_size, void* d_ws, size_t ws_size,
                              hipStream_t stream) {
    const float* X  = (const float*)d_in[0];
    const float* wq = (const float*)d_in[1];
    const float* bq = (const float*)d_in[2];
    const float* wk = (const float*)d_in[3];
    const float* bk = (const float*)d_in[4];
    const float* wv = (const float*)d_in[5];
    const float* bv = (const float*)d_in[6];
    float* out = (float*)d_out;

    // workspace layout
    const size_t offW   = 0;                                  // 3072*1024 bf16
    const size_t offB   = offW + (size_t)3072 * 1024 * 2;     // 3072 f32
    const size_t offQKV = offB + (size_t)3072 * 4;            // 20992*3072 bf16
    const size_t total  = offQKV + (size_t)20992 * 3072 * 2;  // ~135 MB
    if (ws_size < total) return;

    u16*   wcat = (u16*)  ((char*)d_ws + offW);
    float* bcat = (float*)((char*)d_ws + offB);
    u16*   qkv  = (u16*)  ((char*)d_ws + offQKV);

    convert_weights<<<3072, 256, 0, stream>>>(wq, wk, wv, bq, bk, bv, wcat, bcat);
    gemm_qkv<<<dim3(20992 / BM, 3072 / BN), 256, 0, stream>>>(X, wcat, bcat, qkv);
    attention<<<512 * 16, 32, 0, stream>>>(qkv, out);
}
